// MultiHeadAttention_60859686584499
// MI455X (gfx1250) — matrix operand order repaired and checked by reading, not compiled, not run on hardware
//
#include <hip/hip_runtime.h>
#include <hip/hip_bf16.h>

// ---------------------------------------------------------------------------
// MHA forward for MI455X (gfx1250, wave32, WMMA 16x16x32 f16)
//  - all GEMMs via v_wmma_f32_16x16x32_f16
//  - dense GEMMs register-blocked 32x64 per wave (8 WMMA / K-step)
//  - flash attention with online softmax, LDS layout round-trip for P
// ---------------------------------------------------------------------------

typedef __attribute__((ext_vector_type(16))) _Float16 v16h;
typedef __attribute__((ext_vector_type(8)))  _Float16 v8h;
typedef __attribute__((ext_vector_type(8)))  float    v8f;

union AFrag { v16h v; v8h h[2]; };

__device__ __forceinline__ v8f wmma_f16(v16h a, v16h b, v8f c) {
  // D = A(16x32 f16) * B(32x16 f16) + C(16x16 f32)
  return __builtin_amdgcn_wmma_f32_16x16x32_f16(
      /*neg_a=*/false, a, /*neg_b=*/false, b,
      /*c_mod=*/(short)0, c, /*reuse_a=*/false, /*reuse_b=*/false);
}

namespace cfg {
constexpr int B  = 2;
constexpr int T  = 2048;
constexpr int C  = 1024;
constexpr int H  = 16;
constexpr int DH = 64;
constexpr int HD = H * DH;       // 1024
constexpr int BT = B * T;        // 4096
}

// ---------------------------------------------------------------------------
// Kernel 1: fp32 -> f16 elementwise
// ---------------------------------------------------------------------------
__global__ void mha_cvt_f16(const float* __restrict__ in,
                            _Float16* __restrict__ out, int n) {
  int i = blockIdx.x * blockDim.x + threadIdx.x;
  if (i < n) out[i] = (_Float16)in[i];
}

// ---------------------------------------------------------------------------
// Kernel 2: per-head weight transpose (H,C,DH) f32 -> (H,DH,C) f16
// ---------------------------------------------------------------------------
__global__ void mha_tr_head(const float* __restrict__ in,
                            _Float16* __restrict__ out) {
  using namespace cfg;
  int i = blockIdx.x * blockDim.x + threadIdx.x;
  int n = H * C * DH;
  if (i >= n) return;
  int h = i / (C * DH);
  int r = i - h * (C * DH);
  int c = r / DH;
  int d = r - c * DH;
  out[(size_t)h * DH * C + (size_t)d * C + c] = (_Float16)in[i];
}

// ---------------------------------------------------------------------------
// Kernel 3: Wp transpose (HD,C) f32 -> (C,HD) f16
// ---------------------------------------------------------------------------
__global__ void mha_tr_wp(const float* __restrict__ in,
                          _Float16* __restrict__ out) {
  using namespace cfg;
  int i = blockIdx.x * blockDim.x + threadIdx.x;
  int n = HD * C;
  if (i >= n) return;
  int k = i / C;
  int c = i - k * C;
  out[(size_t)c * HD + k] = (_Float16)in[i];
}

// ---------------------------------------------------------------------------
// Kernel 4: QKV projection, register-blocked 32x64 per wave.
//   Q[b,h] = x[b] (TxC) @ Wq[h] (CxDH)   (same for K,V)
//   Q,K stored (B,H,T,DH) f16 ; V stored transposed (B,H,DH,T) f16.
//   grid = (T/128, 3*H, B), block = 128 (4 waves, 32 rows x 64 cols each)
// ---------------------------------------------------------------------------
__global__ void mha_qkv_gemm(const _Float16* __restrict__ xh,
                             const _Float16* __restrict__ WqT,
                             const _Float16* __restrict__ WkT,
                             const _Float16* __restrict__ WvT,
                             _Float16* __restrict__ Qh,
                             _Float16* __restrict__ Kh,
                             _Float16* __restrict__ VhT) {
  using namespace cfg;
  const int lane  = threadIdx.x & 31;
  const int wave  = threadIdx.x >> 5;
  const int mBase = blockIdx.x * 128 + wave * 32;   // 32 rows per wave
  const int hm    = blockIdx.y;                     // 0..3H-1
  const int b     = blockIdx.z;
  const int h     = hm % H;
  const int which = hm / H;                         // 0=Q 1=K 2=V

  const _Float16* WT = (which == 0) ? WqT : (which == 1) ? WkT : WvT;
  WT += (size_t)h * DH * C;

  const int aSel = (lane >> 4) * 16;                // A K-group select
  const int bSel = (lane >> 4) * 16;                // B K-group select

  const _Float16* aRow0 = xh + (size_t)b * T * C + (size_t)(mBase + (lane & 15)) * C;
  const _Float16* aRow1 = aRow0 + (size_t)16 * C;
  const _Float16* bRowB = WT + (size_t)(lane & 15) * C;

  v8f acc[2][4] = {};
  for (int k0 = 0; k0 < C; k0 += 32) {
    AFrag a0, a1;
    a0.h[0] = *(const v8h*)(aRow0 + k0 + aSel);
    a0.h[1] = *(const v8h*)(aRow0 + k0 + aSel + 8);
    a1.h[0] = *(const v8h*)(aRow1 + k0 + aSel);
    a1.h[1] = *(const v8h*)(aRow1 + k0 + aSel + 8);
    __builtin_prefetch(aRow0 + k0 + 128, 0, 1);     // global_prefetch_b8
    __builtin_prefetch(aRow1 + k0 + 128, 0, 1);
#pragma unroll
    for (int j = 0; j < 4; ++j) {
      v16h bf = *(const v16h*)(bRowB + (size_t)j * 16 * C + k0 + bSel);
      acc[0][j] = wmma_f16(a0.v, bf, acc[0][j]);
      acc[1][j] = wmma_f16(a1.v, bf, acc[1][j]);
    }
  }

  const size_t bh = (size_t)b * H + h;
  if (which < 2) {
    _Float16* out = ((which == 0) ? Qh : Kh) + bh * T * DH;
#pragma unroll
    for (int mt = 0; mt < 2; ++mt)
#pragma unroll
      for (int j = 0; j < 4; ++j)
#pragma unroll
        for (int i = 0; i < 8; ++i) {
          int r = mBase + mt * 16 + i + ((lane >> 4) << 3);
          int d = j * 16 + (lane & 15);
          out[(size_t)r * DH + d] = (_Float16)acc[mt][j][i];
        }
  } else {
    _Float16* out = VhT + bh * DH * T;               // (DH, T)
#pragma unroll
    for (int mt = 0; mt < 2; ++mt)
#pragma unroll
      for (int j = 0; j < 4; ++j)
#pragma unroll
        for (int i = 0; i < 8; ++i) {
          int r = mBase + mt * 16 + i + ((lane >> 4) << 3);
          int d = j * 16 + (lane & 15);
          out[(size_t)d * T + r] = (_Float16)acc[mt][j][i];
        }
  }
}

// ---------------------------------------------------------------------------
// Kernel 5: flash attention, one wave per 16-query tile.
//   grid = (T/64, H, B), block = 128 (4 waves)
//   writes AO (B*T, H*DH) f16
// ---------------------------------------------------------------------------
__global__ void mha_attn(const _Float16* __restrict__ Qh,
                         const _Float16* __restrict__ Kh,
                         const _Float16* __restrict__ VhT,
                         const int* __restrict__ lengths,
                         _Float16* __restrict__ AO) {
  using namespace cfg;
  const int lane  = threadIdx.x & 31;
  const int wave  = threadIdx.x >> 5;
  const int qTile = blockIdx.x * 4 + wave;
  const int h     = blockIdx.y;
  const int b     = blockIdx.z;
  const size_t bh = (size_t)b * H + h;

  const _Float16* Qp = Qh  + bh * T * DH;   // (T, DH)
  const _Float16* Kp = Kh  + bh * T * DH;   // (T, DH)
  const _Float16* Vp = VhT + bh * DH * T;   // (DH, T)
  const int len = lengths[b];
  const float scale = 0.125f;               // 1/sqrt(64)

  __shared__ alignas(16) _Float16 pb[4][16 * 32];

  // Q fragments for K-chunks d=[0,32) and d=[32,64)
  const int rowQ = qTile * 16 + (lane & 15);
  const int aSel = (lane >> 4) * 16;
  AFrag qf[2];
#pragma unroll
  for (int c = 0; c < 2; ++c) {
    const _Float16* p = Qp + (size_t)rowQ * DH + c * 32 + aSel;
    qf[c].h[0] = *(const v8h*)p;
    qf[c].h[1] = *(const v8h*)(p + 8);
  }

  v8f o0 = {}, o1 = {}, o2 = {}, o3 = {};
  float m[8], l[8];
#pragma unroll
  for (int i = 0; i < 8; ++i) { m[i] = -1e30f; l[i] = 0.f; }

  int sEnd = (len + 31) & ~31;
  if (sEnd > T) sEnd = T;
  const int bSel = (lane >> 4) * 16;

  for (int s0 = 0; s0 < sEnd; s0 += 32) {
    // ---- scores: S = Q @ K^T (two 16-key tiles) ----
    v8f S0 = {}, S1 = {};
    const _Float16* k0p = Kp + (size_t)(s0 + (lane & 15)) * DH;
    const _Float16* k1p = Kp + (size_t)(s0 + 16 + (lane & 15)) * DH;
#pragma unroll
    for (int c = 0; c < 2; ++c) {
      v16h kb0 = *(const v16h*)(k0p + c * 32 + bSel);
      S0 = wmma_f16(qf[c].v, kb0, S0);
      v16h kb1 = *(const v16h*)(k1p + c * 32 + bSel);
      S1 = wmma_f16(qf[c].v, kb1, S1);
    }

    // ---- mask + online softmax (rows live across 16-lane groups) ----
    const bool v0 = (s0 + (lane & 15)) < len;
    const bool v1 = (s0 + 16 + (lane & 15)) < len;
#pragma unroll
    for (int i = 0; i < 8; ++i) {
      float x0 = v0 ? S0[i] * scale : -1e30f;
      float x1 = v1 ? S1[i] * scale : -1e30f;
      float mx = fmaxf(x0, x1);
#pragma unroll
      for (int off = 1; off < 16; off <<= 1)
        mx = fmaxf(mx, __shfl_xor(mx, off, 16));
      float mn = fmaxf(m[i], mx);
      float al = __expf(m[i] - mn);
      l[i] *= al;
      o0[i] *= al; o1[i] *= al; o2[i] *= al; o3[i] *= al;
      float p0 = __expf(x0 - mn);
      float p1 = __expf(x1 - mn);
      float rs = p0 + p1;
#pragma unroll
      for (int off = 1; off < 16; off <<= 1)
        rs += __shfl_xor(rs, off, 16);
      l[i] += rs;
      m[i] = mn;
      // stash P tile (C/D layout -> LDS row-major 16x32)
      int r = i + ((lane >> 4) << 3);
      pb[wave][r * 32 + (lane & 15)]      = (_Float16)p0;
      pb[wave][r * 32 + 16 + (lane & 15)] = (_Float16)p1;
    }

    asm volatile("s_wait_dscnt 0" ::: "memory");

    // ---- reload P in A-fragment layout (16x32, K=32) ----
    AFrag pf;
    {
      const int rq = lane & 15;
      pf.h[0] = *(const v8h*)&pb[wave][rq * 32 + aSel];
      pf.h[1] = *(const v8h*)&pb[wave][rq * 32 + aSel + 8];
    }

    // ---- O += P @ V  (V^T stored (DH,T): contiguous B fragments) ----
    const int kSel = s0 + ((lane >> 4) << 4);
    v16h vb;
    vb = *(const v16h*)(Vp + (size_t)( 0 + (lane & 15)) * T + kSel);
    o0 = wmma_f16(pf.v, vb, o0);
    vb = *(const v16h*)(Vp + (size_t)(16 + (lane & 15)) * T + kSel);
    o1 = wmma_f16(pf.v, vb, o1);
    vb = *(const v16h*)(Vp + (size_t)(32 + (lane & 15)) * T + kSel);
    o2 = wmma_f16(pf.v, vb, o2);
    vb = *(const v16h*)(Vp + (size_t)(48 + (lane & 15)) * T + kSel);
    o3 = wmma_f16(pf.v, vb, o3);
  }

  // ---- normalize, zero padded queries, store AO as f16 ----
  _Float16* aoRowBase = AO + ((size_t)b * T) * HD;
#pragma unroll
  for (int i = 0; i < 8; ++i) {
    int rloc = i + ((lane >> 4) << 3);
    int t = qTile * 16 + rloc;
    float sc2 = (l[i] > 0.f) ? (1.0f / l[i]) : 0.f;
    if (t >= len) sc2 = 0.f;
    _Float16* row = aoRowBase + (size_t)t * HD + h * DH + (lane & 15);
    row[0]  = (_Float16)(o0[i] * sc2);
    row[16] = (_Float16)(o1[i] * sc2);
    row[32] = (_Float16)(o2[i] * sc2);
    row[48] = (_Float16)(o3[i] * sc2);
  }
}

// ---------------------------------------------------------------------------
// Kernel 6: output projection  out = AO (BT x HD) @ Wp (HD x C) + bp, fp32
//   register-blocked 32x64 per wave
//   grid = (BT/128, C/64), block = 128
// ---------------------------------------------------------------------------
__global__ void mha_proj(const _Float16* __restrict__ AO,
                         const _Float16* __restrict__ WpT,  // (C, HD)
                         const float* __restrict__ bp,
                         float* __restrict__ out) {
  using namespace cfg;
  const int lane  = threadIdx.x & 31;
  const int wave  = threadIdx.x >> 5;
  const int mBase = blockIdx.x * 128 + wave * 32;
  const int cBase = blockIdx.y * 64;
  const int aSel  = (lane >> 4) * 16;
  const int bSel  = (lane >> 4) * 16;

  const _Float16* aRow0 = AO + (size_t)(mBase + (lane & 15)) * HD;
  const _Float16* aRow1 = aRow0 + (size_t)16 * HD;
  const _Float16* bRowB = WpT + (size_t)(cBase + (lane & 15)) * HD;

  v8f acc[2][4] = {};
  for (int k0 = 0; k0 < HD; k0 += 32) {
    AFrag a0, a1;
    a0.h[0] = *(const v8h*)(aRow0 + k0 + aSel);
    a0.h[1] = *(const v8h*)(aRow0 + k0 + aSel + 8);
    a1.h[0] = *(const v8h*)(aRow1 + k0 + aSel);
    a1.h[1] = *(const v8h*)(aRow1 + k0 + aSel + 8);
    __builtin_prefetch(bRowB + k0 + 128, 0, 1);
#pragma unroll
    for (int j = 0; j < 4; ++j) {
      v16h bf = *(const v16h*)(bRowB + (size_t)j * 16 * HD + k0 + bSel);
      acc[0][j] = wmma_f16(a0.v, bf, acc[0][j]);
      acc[1][j] = wmma_f16(a1.v, bf, acc[1][j]);
    }
  }

#pragma unroll
  for (int mt = 0; mt < 2; ++mt)
#pragma unroll
    for (int j = 0; j < 4; ++j) {
      int col = cBase + j * 16 + (lane & 15);
      float bias = bp[col];
#pragma unroll
      for (int i = 0; i < 8; ++i) {
        int r = mBase + mt * 16 + i + ((lane >> 4) << 3);
        out[(size_t)r * C + col] = acc[mt][j][i] + bias;
      }
    }
}

// ---------------------------------------------------------------------------
// Launch
// ---------------------------------------------------------------------------
extern "C" void kernel_launch(void* const* d_in, const int* in_sizes, int n_in,
                              void* d_out, int out_size, void* d_ws, size_t ws_size,
                              hipStream_t stream) {
  using namespace cfg;
  const float* x       = (const float*)d_in[0];
  const int*   lengths = (const int*)  d_in[1];
  const float* Wq      = (const float*)d_in[2];
  const float* Wk      = (const float*)d_in[3];
  const float* Wv      = (const float*)d_in[4];
  const float* Wp      = (const float*)d_in[5];
  const float* bp      = (const float*)d_in[6];
  float* out = (float*)d_out;

  // workspace carve-up (f16 buffers)
  char* w = (char*)d_ws;
  const size_t nX  = (size_t)B * T * C;        // 4M
  const size_t nW  = (size_t)H * C * DH;       // 1M
  const size_t nWp = (size_t)HD * C;           // 1M
  const size_t nQ  = (size_t)B * H * T * DH;   // 4M
  const size_t nAO = (size_t)BT * HD;          // 4M

  _Float16* xh  = (_Float16*)w;                 w += nX  * 2;
  _Float16* WqT = (_Float16*)w;                 w += nW  * 2;
  _Float16* WkT = (_Float16*)w;                 w += nW  * 2;
  _Float16* WvT = (_Float16*)w;                 w += nW  * 2;
  _Float16* WpT = (_Float16*)w;                 w += nWp * 2;
  _Float16* Qh  = (_Float16*)w;                 w += nQ  * 2;
  _Float16* Kh  = (_Float16*)w;                 w += nQ  * 2;
  _Float16* VhT = (_Float16*)w;                 w += nQ  * 2;
  _Float16* AO  = (_Float16*)w;                 w += nAO * 2;

  // 1) convert x
  mha_cvt_f16<<<(int)((nX + 255) / 256), 256, 0, stream>>>(x, xh, (int)nX);
  // 2) transpose weights
  mha_tr_head<<<(int)((nW + 255) / 256), 256, 0, stream>>>(Wq, WqT);
  mha_tr_head<<<(int)((nW + 255) / 256), 256, 0, stream>>>(Wk, WkT);
  mha_tr_head<<<(int)((nW + 255) / 256), 256, 0, stream>>>(Wv, WvT);
  mha_tr_wp  <<<(int)((nWp + 255) / 256), 256, 0, stream>>>(Wp, WpT);
  // 3) QKV projection (WMMA, 32x64 per wave)
  mha_qkv_gemm<<<dim3(T / 128, 3 * H, B), 128, 0, stream>>>(
      xh, WqT, WkT, WvT, Qh, Kh, VhT);
  // 4) flash attention (WMMA)
  mha_attn<<<dim3(T / 64, H, B), 128, 0, stream>>>(Qh, Kh, VhT, lengths, AO);
  // 5) output projection (WMMA, 32x64 per wave)
  mha_proj<<<dim3(BT / 128, C / 64), 128, 0, stream>>>(AO, WpT, bp, out);
}
